// DNA_5841155523070
// MI455X (gfx1250) — compile-verified, matrix operand order checked by reading, not run on hardware
//
#include <hip/hip_runtime.h>
#include <hip/hip_bf16.h>

// Problem constants (fixed by the reference).
#define TT   4096
#define DD   1024
#define EE   16
#define DFF  1024
#define CAPC 512

typedef __bf16 bf16_t;
typedef bf16_t v16bf __attribute__((ext_vector_type(16)));
typedef float  v8f   __attribute__((ext_vector_type(8)));
typedef unsigned int v8u __attribute__((ext_vector_type(8)));

__device__ __forceinline__ unsigned short f2bf(float x) {
    // round-to-nearest-even fp32 -> bf16
    unsigned int b = __float_as_uint(x);
    unsigned int r = b + 0x7FFFu + ((b >> 16) & 1u);
    return (unsigned short)(r >> 16);
}

__device__ __forceinline__ float gelu_tanh(float x) {
    float x3 = x * x * x;
    float t = tanhf(0.7978845608028654f * (x + 0.044715f * x3));
    return 0.5f * x * (1.0f + t);
}

__device__ __forceinline__ v8f wmma_bf16(v16bf a, v16bf b, v8f c) {
#if defined(__AMDGCN__)
    return __builtin_amdgcn_wmma_f32_16x16x32_bf16(
        /*neg_a=*/false, a, /*neg_b=*/false, b,
        /*c_mod=*/(short)0, c, /*reuse_a=*/false, /*reuse_b=*/false);
#else
    (void)a; (void)b;
    return c;
#endif
}

// CDNA5 async global->LDS copy (16B per lane), tracked by ASYNCcnt.
// LDS address comes from the vdst VGPR; generic LDS addr low 32 bits == LDS offset.
__device__ __forceinline__ void async_copy16(void* lds_ptr, const void* gptr) {
#if defined(__AMDGCN__)
    unsigned lds_off = (unsigned)(unsigned long long)lds_ptr;
    asm volatile("global_load_async_to_lds_b128 %0, %1, off"
                 :: "v"(lds_off), "v"(gptr)
                 : "memory");
#else
    (void)lds_ptr; (void)gptr;
#endif
}

__device__ __forceinline__ void wait_async0() {
#if defined(__AMDGCN__)
    asm volatile("s_wait_asynccnt 0x0" ::: "memory");
#endif
}

// ---------------------------------------------------------------------------
// 1) Router: logits = h @ Wr, softmax probs, top-2 expert ids per token.
//    One block per token; 16 waves, one per expert column; wave32 reduction.
// ---------------------------------------------------------------------------
__global__ __launch_bounds__(512)
void router_kernel(const float* __restrict__ h, const float* __restrict__ Wr,
                   float* __restrict__ logits, float* __restrict__ probs,
                   int* __restrict__ top2) {
    const int t    = blockIdx.x;
    const int wave = threadIdx.x >> 5;   // expert id, 0..15
    const int lane = threadIdx.x & 31;
    const float* hr = h + (size_t)t * DD;

    float p = 0.f;
    for (int d = lane; d < DD; d += 32)
        p += hr[d] * Wr[(size_t)d * EE + wave];
    for (int off = 16; off > 0; off >>= 1)
        p += __shfl_xor(p, off, 32);

    __shared__ float lg[EE];
    if (lane == 0) lg[wave] = p;
    __syncthreads();

    if (threadIdx.x == 0) {
        float mx = lg[0];
        for (int e = 1; e < EE; e++) mx = fmaxf(mx, lg[e]);
        float ex[EE]; float s = 0.f;
        for (int e = 0; e < EE; e++) { ex[e] = __expf(lg[e] - mx); s += ex[e]; }
        float inv = 1.f / s;
        // top-2 with jax.lax.top_k tie-breaking (strictly greater keeps lower idx)
        int b0 = 0; float v0 = lg[0];
        for (int e = 1; e < EE; e++) if (lg[e] > v0) { v0 = lg[e]; b0 = e; }
        int b1 = -1; float v1 = -INFINITY;
        for (int e = 0; e < EE; e++) if (e != b0 && lg[e] > v1) { v1 = lg[e]; b1 = e; }
        for (int e = 0; e < EE; e++) {
            logits[(size_t)t * EE + e] = lg[e];
            probs [(size_t)t * EE + e] = ex[e] * inv;
        }
        top2[t * 2 + 0] = b0;
        top2[t * 2 + 1] = b1;
    }
}

// ---------------------------------------------------------------------------
// 2) Capacity selection: per expert, top-C tokens by logit (ties -> lower
//    token id), then slots ordered by token position. Exactly reproduces the
//    reference's top_k + stable argsort via a full bitonic sort on 64-bit keys
//    key = (~ord(score) << 32) | token  (ascending sort == descending score).
// ---------------------------------------------------------------------------
__global__ __launch_bounds__(1024)
void capacity_kernel(const float* __restrict__ logits, const int* __restrict__ top2,
                     int* __restrict__ slot_token, int* __restrict__ slot_of) {
    const int e   = blockIdx.x;
    const int tid = threadIdx.x;

    __shared__ unsigned long long keys[TT];   // 32 KB
    __shared__ unsigned int       pos[CAPC];  // 2 KB

    for (int i = tid; i < TT; i += 1024) {
        bool m = (top2[i * 2] == e) || (top2[i * 2 + 1] == e);
        unsigned int desc;
        if (m) {
            unsigned int b   = __float_as_uint(logits[(size_t)i * EE + e]);
            unsigned int asc = (b & 0x80000000u) ? ~b : (b | 0x80000000u);
            desc = ~asc;
        } else {
            desc = 0xFF800000u;  // maps -inf: sorts after every finite logit
        }
        keys[i] = ((unsigned long long)desc << 32) | (unsigned int)i;
    }
    __syncthreads();

    // Bitonic sort, ascending on key -> descending score, ties ascending token.
    for (int k = 2; k <= TT; k <<= 1) {
        for (int j = k >> 1; j > 0; j >>= 1) {
            for (int i = tid; i < TT; i += 1024) {
                int ixj = i ^ j;
                if (ixj > i) {
                    unsigned long long a = keys[i], b = keys[ixj];
                    bool up = ((i & k) == 0);
                    if ((a > b) == up) { keys[i] = b; keys[ixj] = a; }
                }
            }
            __syncthreads();
        }
    }

    // Top-C slots; valid iff the score half isn't the -inf sentinel.
    if (tid < CAPC) {
        unsigned long long kk = keys[tid];
        bool valid = ((unsigned int)(kk >> 32)) != 0xFF800000u;
        pos[tid] = valid ? (unsigned int)kk : 0x7FFFFFFFu;
    }
    __syncthreads();

    // Stable-by-position ordering == sort selected token ids ascending,
    // invalid sentinels sink to the end.
    for (int k = 2; k <= CAPC; k <<= 1) {
        for (int j = k >> 1; j > 0; j >>= 1) {
            if (tid < CAPC) {
                int i = tid;
                int ixj = i ^ j;
                if (ixj > i) {
                    unsigned int a = pos[i], b = pos[ixj];
                    bool up = ((i & k) == 0);
                    if ((a > b) == up) { pos[i] = b; pos[ixj] = a; }
                }
            }
            __syncthreads();
        }
    }

    for (int i = tid; i < TT; i += 1024) slot_of[(size_t)e * TT + i] = -1;
    __syncthreads();
    if (tid < CAPC) {
        unsigned int p = pos[tid];
        int tok = (p == 0x7FFFFFFFu) ? -1 : (int)p;
        slot_token[e * CAPC + tid] = tok;
        if (tok >= 0) slot_of[(size_t)e * TT + tok] = tid;
    }
}

// ---------------------------------------------------------------------------
// 3) Dispatch: xin[e][c][:] = bf16(h[token])  (zeros for invalid slots)
// ---------------------------------------------------------------------------
__global__ __launch_bounds__(256)
void dispatch_kernel(const float* __restrict__ h, const int* __restrict__ slot_token,
                     unsigned short* __restrict__ xin) {
    const int ec  = blockIdx.x;           // e*CAP + c
    const int tok = slot_token[ec];
    unsigned short* dst = xin + (size_t)ec * DD;
    const int d = threadIdx.x * 4;
    ushort4 o = {0, 0, 0, 0};
    if (tok >= 0) {
        const float4 v = *(const float4*)(h + (size_t)tok * DD + d);
        o.x = f2bf(v.x); o.y = f2bf(v.y); o.z = f2bf(v.z); o.w = f2bf(v.w);
    }
    *(ushort4*)(dst + d) = o;
}

// ---------------------------------------------------------------------------
// 4/5) Tiled WMMA GEMM, per expert:  Cmat[M=512,N=1024] = A[M,K=1024] * B[K,N]
//    A: bf16 row-major in global, staged with CDNA5 async global->LDS copies.
//    B: fp32 row-major, converted to bf16 + transposed on the way into LDS.
//    Double-buffered LDS: stage tile k+1 while computing tile k.
//    Block tile 128x128, 8 waves, each wave a 64x32 tile = 4x2 WMMA tiles.
//    MODE 0: GELU + bf16 output (hidden).  MODE 1: f32 output (expert_out).
// ---------------------------------------------------------------------------
template <int MODE>
__global__ __launch_bounds__(256)
void gemm_kernel(const unsigned short* __restrict__ Aall,
                 const float* __restrict__ Ball, void* __restrict__ Call) {
    constexpr int M = CAPC, N = 1024, K = 1024;
    const int e  = blockIdx.z;
    const int m0 = blockIdx.y * 128;
    const int n0 = blockIdx.x * 128;
    const unsigned short* A = Aall + (size_t)e * M * K;
    const float*          B = Ball + (size_t)e * K * N;

    // 48-ushort row stride => 96 B, rows stay 16-B aligned for ds_load_b128.
    __shared__ __align__(16) unsigned short As[2][128][48];  // [m][k]
    __shared__ __align__(16) unsigned short Bs[2][128][48];  // [n][k] (transposed)

    const int tid  = threadIdx.x;
    const int wave = tid >> 5, lane = tid & 31;
    const int wm = wave & 1, wn = wave >> 1;   // 2x4 wave grid
    const int hl = lane >> 4, l16 = lane & 15; // half-select + row/col in frag

    // Fixed staging coordinates.
    const int ar  = tid >> 2;            // A rows 0..63 (and +64), chunk (tid&3)*8
    const int ak  = (tid & 3) * 8;
    const int bkk = tid >> 5;            // B k-rows 0..7 (and +8,+16,+24)
    const int bn4 = (tid & 31) * 4;      // B n-cols, 4 at a time

    v8f acc[4][2];
    for (int tm = 0; tm < 4; tm++)
        for (int tn = 0; tn < 2; tn++)
            for (int r = 0; r < 8; r++) acc[tm][tn][r] = 0.f;

    const int kb0 = hl * 8;        // K offsets per the 16-bit fragment layout
    const int kb1 = 16 + hl * 8;

    // Stage one 128x32 K-slab (A async -> LDS, B load/convert/transpose).
    auto stage = [&](int buf, int k0) {
        async_copy16(&As[buf][ar][ak],      A + (size_t)(m0 + ar) * K + k0 + ak);
        async_copy16(&As[buf][64 + ar][ak], A + (size_t)(m0 + 64 + ar) * K + k0 + ak);

        const float4 v0 = *(const float4*)(B + (size_t)(k0 + bkk +  0) * N + n0 + bn4);
        const float4 v1 = *(const float4*)(B + (size_t)(k0 + bkk +  8) * N + n0 + bn4);
        const float4 v2 = *(const float4*)(B + (size_t)(k0 + bkk + 16) * N + n0 + bn4);
        const float4 v3 = *(const float4*)(B + (size_t)(k0 + bkk + 24) * N + n0 + bn4);
        Bs[buf][bn4 + 0][bkk +  0] = f2bf(v0.x);
        Bs[buf][bn4 + 1][bkk +  0] = f2bf(v0.y);
        Bs[buf][bn4 + 2][bkk +  0] = f2bf(v0.z);
        Bs[buf][bn4 + 3][bkk +  0] = f2bf(v0.w);
        Bs[buf][bn4 + 0][bkk +  8] = f2bf(v1.x);
        Bs[buf][bn4 + 1][bkk +  8] = f2bf(v1.y);
        Bs[buf][bn4 + 2][bkk +  8] = f2bf(v1.z);
        Bs[buf][bn4 + 3][bkk +  8] = f2bf(v1.w);
        Bs[buf][bn4 + 0][bkk + 16] = f2bf(v2.x);
        Bs[buf][bn4 + 1][bkk + 16] = f2bf(v2.y);
        Bs[buf][bn4 + 2][bkk + 16] = f2bf(v2.z);
        Bs[buf][bn4 + 3][bkk + 16] = f2bf(v2.w);
        Bs[buf][bn4 + 0][bkk + 24] = f2bf(v3.x);
        Bs[buf][bn4 + 1][bkk + 24] = f2bf(v3.y);
        Bs[buf][bn4 + 2][bkk + 24] = f2bf(v3.z);
        Bs[buf][bn4 + 3][bkk + 24] = f2bf(v3.w);
    };

    stage(0, 0);
    wait_async0();
    __syncthreads();

    int buf = 0;
    for (int k0 = 0; k0 < K; k0 += 32, buf ^= 1) {
        if (k0 + 32 < K) stage(buf ^ 1, k0 + 32);   // overlap with compute below

        v16bf afrag[4], bfrag[2];
        for (int tm = 0; tm < 4; tm++) {
            int row = wm * 64 + tm * 16 + l16;
            uint4 lo = *(const uint4*)&As[buf][row][kb0];
            uint4 hi = *(const uint4*)&As[buf][row][kb1];
            v8u u; u[0]=lo.x; u[1]=lo.y; u[2]=lo.z; u[3]=lo.w;
                   u[4]=hi.x; u[5]=hi.y; u[6]=hi.z; u[7]=hi.w;
            afrag[tm] = __builtin_bit_cast(v16bf, u);
        }
        for (int tn = 0; tn < 2; tn++) {
            int col = wn * 32 + tn * 16 + l16;
            uint4 lo = *(const uint4*)&Bs[buf][col][kb0];
            uint4 hi = *(const uint4*)&Bs[buf][col][kb1];
            v8u u; u[0]=lo.x; u[1]=lo.y; u[2]=lo.z; u[3]=lo.w;
                   u[4]=hi.x; u[5]=hi.y; u[6]=hi.z; u[7]=hi.w;
            bfrag[tn] = __builtin_bit_cast(v16bf, u);
        }
        for (int tm = 0; tm < 4; tm++)
            for (int tn = 0; tn < 2; tn++)
                acc[tm][tn] = wmma_bf16(afrag[tm], bfrag[tn], acc[tm][tn]);

        wait_async0();      // next-slab async A copies landed
        __syncthreads();    // next-slab B ds stores visible; all reads of buf done
    }

    // Epilogue: C/D layout -> lanes<16 own M=r, lanes>=16 own M=8+r, N=lane&15.
    for (int tm = 0; tm < 4; tm++) {
        for (int tn = 0; tn < 2; tn++) {
            const int col  = n0 + wn * 32 + tn * 16 + l16;
            const int rowb = m0 + wm * 64 + tm * 16 + (hl ? 8 : 0);
            for (int r = 0; r < 8; r++) {
                float v = acc[tm][tn][r];
                size_t idx = (size_t)e * M * N + (size_t)(rowb + r) * N + col;
                if (MODE == 0) {
                    ((unsigned short*)Call)[idx] = f2bf(gelu_tanh(v));
                } else {
                    ((float*)Call)[idx] = v;
                }
            }
        }
    }
}

// ---------------------------------------------------------------------------
// 6) Combine: out[t] = h[t]*(1-rho) + sum over kept top-2 experts of
//    probs[t][e] * expert_out[e][slot]. Fixed order -> deterministic.
// ---------------------------------------------------------------------------
__global__ __launch_bounds__(256)
void combine_kernel(const float* __restrict__ h, const float* __restrict__ probs,
                    const int* __restrict__ top2, const int* __restrict__ slot_of,
                    const float* __restrict__ eout, float* __restrict__ out) {
    const int t  = blockIdx.x;
    const int e0 = top2[t * 2 + 0];
    const int e1 = top2[t * 2 + 1];
    const int s0 = slot_of[(size_t)e0 * TT + t];
    const int s1 = slot_of[(size_t)e1 * TT + t];
    const float w0 = (s0 >= 0) ? probs[(size_t)t * EE + e0] : 0.f;
    const float w1 = (s1 >= 0) ? probs[(size_t)t * EE + e1] : 0.f;
    const float keep = 1.f - (w0 + w1);

    const int d = threadIdx.x * 4;
    float4 hv = *(const float4*)(h + (size_t)t * DD + d);
    float ax = hv.x * keep, ay = hv.y * keep, az = hv.z * keep, aw = hv.w * keep;
    if (s0 >= 0) {
        const float4 v = *(const float4*)(eout + ((size_t)e0 * CAPC + s0) * DD + d);
        ax += w0 * v.x; ay += w0 * v.y; az += w0 * v.z; aw += w0 * v.w;
    }
    if (s1 >= 0) {
        const float4 v = *(const float4*)(eout + ((size_t)e1 * CAPC + s1) * DD + d);
        ax += w1 * v.x; ay += w1 * v.y; az += w1 * v.z; aw += w1 * v.w;
    }
    float4 r; r.x = ax; r.y = ay; r.z = az; r.w = aw;
    *(float4*)(out + (size_t)t * DD + d) = r;
}

// ---------------------------------------------------------------------------
// Host launcher
// ---------------------------------------------------------------------------
extern "C" void kernel_launch(void* const* d_in, const int* in_sizes, int n_in,
                              void* d_out, int out_size, void* d_ws, size_t ws_size,
                              hipStream_t stream) {
    (void)in_sizes; (void)n_in; (void)out_size; (void)ws_size;

    const float* h  = (const float*)d_in[0];  // [T, D]
    const float* Wr = (const float*)d_in[1];  // [D, E]
    const float* W1 = (const float*)d_in[2];  // [E, D, DFF]
    const float* W2 = (const float*)d_in[3];  // [E, DFF, D]
    float* out = (float*)d_out;               // [T, D]

    char* ws = (char*)d_ws;
    size_t off = 0;
    auto alloc = [&](size_t bytes) -> void* {
        void* p = (void*)(ws + off);
        off += (bytes + 255) & ~(size_t)255;
        return p;
    };
    float* logits        = (float*)alloc((size_t)TT * EE * 4);
    float* probs         = (float*)alloc((size_t)TT * EE * 4);
    int*   top2          = (int*)  alloc((size_t)TT * 2 * 4);
    int*   slot_token    = (int*)  alloc((size_t)EE * CAPC * 4);
    int*   slot_of       = (int*)  alloc((size_t)EE * TT * 4);
    unsigned short* xin    = (unsigned short*)alloc((size_t)EE * CAPC * DD  * 2);
    unsigned short* hidden = (unsigned short*)alloc((size_t)EE * CAPC * DFF * 2);
    float* eout          = (float*)alloc((size_t)EE * CAPC * DD * 4);

    router_kernel  <<<TT, 512, 0, stream>>>(h, Wr, logits, probs, top2);
    capacity_kernel<<<EE, 1024, 0, stream>>>(logits, top2, slot_token, slot_of);
    dispatch_kernel<<<EE * CAPC, 256, 0, stream>>>(h, slot_token, xin);

    dim3 ggrid(1024 / 128, CAPC / 128, EE);   // (N tiles, M tiles, experts)
    gemm_kernel<0><<<ggrid, 256, 0, stream>>>(xin, W1, (void*)hidden);
    gemm_kernel<1><<<ggrid, 256, 0, stream>>>(hidden, W2, (void*)eout);

    combine_kernel<<<TT, 256, 0, stream>>>(h, probs, top2, slot_of, eout, out);
}
